// NeighborhoodAggregationEncoder_23776938951296
// MI455X (gfx1250) — matrix operand order repaired and checked by reading, not compiled, not run on hardware
//
#include <hip/hip_runtime.h>
#include <hip/hip_bf16.h>

// ---------------------------------------------------------------------------
// Types / helpers
// ---------------------------------------------------------------------------
typedef __attribute__((ext_vector_type(16))) __bf16 v16bf;
typedef __attribute__((ext_vector_type(8)))  __bf16 v8bf;
typedef __attribute__((ext_vector_type(8)))  float  v8f;
typedef __attribute__((ext_vector_type(4)))  float  v4f;
typedef __attribute__((ext_vector_type(4)))  int    v4i;

__device__ __forceinline__ __bf16 f2bf(float f) {
  unsigned u = __builtin_bit_cast(unsigned, f);
  unsigned r = (u + 0x7FFFu + ((u >> 16) & 1u)) >> 16;   // RNE
  return __builtin_bit_cast(__bf16, (unsigned short)r);
}
__device__ __forceinline__ float bf2f(__bf16 b) {
  unsigned u = ((unsigned)__builtin_bit_cast(unsigned short, b)) << 16;
  return __builtin_bit_cast(float, u);
}

// CDNA5 async global->LDS copy path (ASYNCcnt-tracked, bypasses VGPRs)
#if defined(__gfx1250__) && __has_builtin(__builtin_amdgcn_global_load_async_to_lds_b128)
#define GEMM_ASYNC 1
#else
#define GEMM_ASYNC 0
#endif

// builtin signature (from hipcc diagnostic): (global v4i*, lds v4i*, imm, imm)
typedef __attribute__((address_space(1))) v4i* as1_v4i_p;
typedef __attribute__((address_space(3))) v4i* as3_v4i_p;

__device__ __forceinline__ void wait_async0() {
#if defined(__gfx1250__)
#if __has_builtin(__builtin_amdgcn_s_wait_asynccnt)
  __builtin_amdgcn_s_wait_asynccnt(0);
#else
  asm volatile("s_wait_asynccnt 0x0" ::: "memory");
#endif
#endif
}

// ---------------------------------------------------------------------------
// Convert kernels (run once per buffer; bandwidth-trivial)
// ---------------------------------------------------------------------------
// x (C,N) f32 -> xb (N,C) bf16   (N = 32768 = 2^15)
__global__ __launch_bounds__(256) void k_cvt_xT(const float* __restrict__ x,
                                                __bf16* __restrict__ xb) {
  const long idx = (long)blockIdx.x * 256 + threadIdx.x;   // over C*N
  const int c = (int)(idx >> 15);
  const int n = (int)(idx & 32767);
  xb[(size_t)n * 128 + c] = f2bf(x[idx]);
}

__global__ __launch_bounds__(256) void k_cvt(const float* __restrict__ in,
                                             __bf16* __restrict__ out) {
  const long idx = (long)blockIdx.x * 256 + threadIdx.x;
  out[idx] = f2bf(in[idx]);
}

// in (R, Cc) f32 -> out (Cc, R) bf16
__global__ __launch_bounds__(256) void k_cvt_T(const float* __restrict__ in,
                                               __bf16* __restrict__ out,
                                               int R, int Cc) {
  const long idx = (long)blockIdx.x * 256 + threadIdx.x;   // over R*Cc
  const int r  = (int)(idx / Cc);
  const int c2 = (int)(idx % Cc);
  out[(size_t)c2 * R + r] = f2bf(in[idx]);
}

// ---------------------------------------------------------------------------
// WMMA GEMM:  Y[n][m] = sum_k A[m][k] * B[n][k]  (+bias[m]) (+resid[n][m]) (ReLU)
//   A: (M,KK) bf16 row-major (weights)   -> staged async into LDS per block
//   B: (N,KK) bf16 row-major (activations, voxel-major)
//   out: (N,M) f32 or bf16
// One wave = two 16x16 output tiles (A fragment reused across both WMMAs).
// Block = 8 waves covering 256 voxels x one 16-row m-tile.
// Fragment layouts used here (contiguous k for A and B, not the ISA table order):
//   A (16-bit 16x32): lane<16 -> row m=lane, K 0..15 contiguous; lane>=16 -> K 16..31
//   B (32x16):        lane<16 -> col n=lane, K 0..15 contiguous; lane>=16 -> K 16..31
//   C/D:              lane<16 -> n=lane, m=v (v=0..7); lane>=16 -> n=lane-16, m=8+v
// ---------------------------------------------------------------------------
template <int OUT_BF16, int RELU, int RESID, int KK>
__global__ __launch_bounds__(256) void k_gemm(const __bf16* __restrict__ A,
                                              const __bf16* __restrict__ B,
                                              const float* __restrict__ bias,
                                              const float* __restrict__ resid,
                                              void* __restrict__ outv,
                                              int M, int N) {
  constexpr int LDSROW = KK + 16;            // +32B pad breaks the 256B bank period
  __shared__ __bf16 As[16 * LDSROW];

  const int tid = threadIdx.x;
  const int mtile = blockIdx.y * 16;

  // ---- stage A tile (16 x KK) into LDS: 16B chunk per thread-iteration ----
  {
    const __bf16* Ag = A + (size_t)mtile * KK;
#pragma unroll
    for (int i = 0; i < KK / 128; ++i) {     // 32*KK bytes / 16B / 256 threads
      const int t   = tid + i * 256;         // 16B chunk index (8 bf16)
      const int row = (t * 8) / KK;
      const int col = (t * 8) % KK;
#if GEMM_ASYNC
      __builtin_amdgcn_global_load_async_to_lds_b128(
          (as1_v4i_p)(Ag + t * 8),
          (as3_v4i_p)(As + row * LDSROW + col), 0, 0);
#else
      *(v8bf*)(As + row * LDSROW + col) = *(const v8bf*)(Ag + t * 8);
#endif
    }
#if GEMM_ASYNC
    wait_async0();
#endif
    __syncthreads();
  }

  const int lane  = tid & 31;
  const int wave  = tid >> 5;
  const int ntile = (blockIdx.x * 8 + wave) * 32;
  const int sel   = lane >> 4;
  const int l16   = lane & 15;

  const __bf16* Al  = As + (size_t)l16 * LDSROW;
  const __bf16* Br0 = B + (size_t)(ntile + l16) * KK;
  const __bf16* Br1 = Br0 + (size_t)16 * KK;

  v8f acc0 = {}, acc1 = {};
#pragma unroll
  for (int kb = 0; kb < KK; kb += 32) {
    v8bf alo = *(const v8bf*)(Al + kb + sel * 16);       // ds_load_b128
    v8bf ahi = *(const v8bf*)(Al + kb + sel * 16 + 8);   // ds_load_b128
    v16bf a;
#pragma unroll
    for (int i = 0; i < 8; ++i) { a[i] = alo[i]; a[i + 8] = ahi[i]; }
    v16bf b0 = *(const v16bf*)(Br0 + kb + sel * 16);
    v16bf b1 = *(const v16bf*)(Br1 + kb + sel * 16);
    acc0 = __builtin_amdgcn_wmma_f32_16x16x32_bf16(false, a, false, b0,
                                                   (short)0, acc0, false, false);
    acc1 = __builtin_amdgcn_wmma_f32_16x16x32_bf16(false, a, false, b1,
                                                   (short)0, acc1, false, false);
  }

  const int mb = mtile + sel * 8;
#pragma unroll
  for (int half = 0; half < 2; ++half) {
    const v8f acc = half ? acc1 : acc0;
    const int n   = ntile + l16 + half * 16;
    float r[8];
#pragma unroll
    for (int i = 0; i < 8; ++i) {
      float v = acc[i] + bias[mb + i];
      if (RESID) v += resid[(size_t)n * M + mb + i];
      if (RELU)  v = v > 0.f ? v : 0.f;
      r[i] = v;
    }
    if (OUT_BF16) {
      v8bf o;
#pragma unroll
      for (int i = 0; i < 8; ++i) o[i] = f2bf(r[i]);
      *(v8bf*)((__bf16*)outv + (size_t)n * M + mb) = o;
    } else {
      float* out = (float*)outv + (size_t)n * M + mb;
      v4f lo = {r[0], r[1], r[2], r[3]};
      v4f hi = {r[4], r[5], r[6], r[7]};
      *(v4f*)out       = lo;
      *(v4f*)(out + 4) = hi;
    }
  }
}

// ---------------------------------------------------------------------------
// Neighborhood attention: one thread per (voxel, group). K=3 -> 27 neighbors,
// D=16.  qkv: (N,384) bf16  [q:0..127, k:128..255, v:256..383]
// sam: (N,128) bf16.  h=w=z=32.
// ---------------------------------------------------------------------------
__global__ __launch_bounds__(256) void k_attn(const __bf16* __restrict__ qkv,
                                              const float* __restrict__ rpb,
                                              __bf16* __restrict__ sam) {
  const int t  = blockIdx.x * 256 + threadIdx.x;
  const int g  = t & 7;
  const int n  = t >> 3;
  const int zz = n & 31, ww = (n >> 5) & 31, hh = n >> 10;
  int sh = hh - 1; sh = sh < 0 ? 0 : (sh > 29 ? 29 : sh);
  int sw = ww - 1; sw = sw < 0 ? 0 : (sw > 29 ? 29 : sw);
  int sz = zz - 1; sz = sz < 0 ? 0 : (sz > 29 ? 29 : sz);
  const int qc = g * 16;

  const __bf16* qp = qkv + (size_t)n * 384 + qc;
  float q[16];
  {
    v8bf q0 = *(const v8bf*)(qp);
    v8bf q1 = *(const v8bf*)(qp + 8);
#pragma unroll
    for (int i = 0; i < 8; ++i) { q[i] = bf2f(q0[i]); q[8 + i] = bf2f(q1[i]); }
  }

  const float scale = 0.25f;                 // D^-0.5, D=16
  const float* rp = rpb + g * 125;           // (5,5,5)
  float logit[27];
  float mx = -3.0e38f;
#pragma unroll
  for (int jh = 0; jh < 3; ++jh)
#pragma unroll
    for (int jw = 0; jw < 3; ++jw)
#pragma unroll
      for (int jz = 0; jz < 3; ++jz) {
        const int ih = sh + jh, iw = sw + jw, iz = sz + jz;
        const int nb = (ih << 10) + (iw << 5) + iz;
        const __bf16* kp = qkv + (size_t)nb * 384 + 128 + qc;
        v8bf k0 = *(const v8bf*)(kp);
        v8bf k1 = *(const v8bf*)(kp + 8);
        float acc = 0.f;
#pragma unroll
        for (int i = 0; i < 8; ++i)
          acc += q[i] * bf2f(k0[i]) + q[8 + i] * bf2f(k1[i]);
        const float b = rp[(ih - hh + 2) * 25 + (iw - ww + 2) * 5 + (iz - zz + 2)];
        const int j = (jh * 3 + jw) * 3 + jz;
        logit[j] = acc * scale + b;
        mx = fmaxf(mx, logit[j]);
      }

  float p[27], s = 0.f;
#pragma unroll
  for (int j = 0; j < 27; ++j) { p[j] = __expf(logit[j] - mx); s += p[j]; }
  const float inv = 1.f / s;

  float o[16];
#pragma unroll
  for (int i = 0; i < 16; ++i) o[i] = 0.f;
#pragma unroll
  for (int jh = 0; jh < 3; ++jh)
#pragma unroll
    for (int jw = 0; jw < 3; ++jw)
#pragma unroll
      for (int jz = 0; jz < 3; ++jz) {
        const int nb = ((sh + jh) << 10) + ((sw + jw) << 5) + (sz + jz);
        const float wgt = p[(jh * 3 + jw) * 3 + jz] * inv;
        const __bf16* vp = qkv + (size_t)nb * 384 + 256 + qc;
        v8bf v0 = *(const v8bf*)(vp);
        v8bf v1 = *(const v8bf*)(vp + 8);
#pragma unroll
        for (int i = 0; i < 8; ++i) {
          o[i]     += wgt * bf2f(v0[i]);
          o[8 + i] += wgt * bf2f(v1[i]);
        }
      }

  __bf16* op = sam + (size_t)n * 128 + qc;
  v8bf r0, r1;
#pragma unroll
  for (int i = 0; i < 8; ++i) { r0[i] = f2bf(o[i]); r1[i] = f2bf(o[8 + i]); }
  *(v8bf*)(op)     = r0;
  *(v8bf*)(op + 8) = r1;
}

// ---------------------------------------------------------------------------
// Instance-norm stats: one block per channel, deterministic LDS tree reduce.
// in: (N,C) f32.  stats[2c]=mean, stats[2c+1]=rstd.
// ---------------------------------------------------------------------------
__global__ __launch_bounds__(256) void k_stats(const float* __restrict__ in,
                                               float* __restrict__ stats,
                                               int C, int N) {
  __shared__ float s1[256];
  __shared__ float s2[256];
  const int c = blockIdx.x;
  const int tid = threadIdx.x;
  float a = 0.f, b = 0.f;
  for (int k = tid; k < N; k += 256) {
    float v = in[(size_t)k * C + c];
    a += v; b += v * v;
  }
  s1[tid] = a; s2[tid] = b;
  __syncthreads();
  for (int off = 128; off > 0; off >>= 1) {
    if (tid < off) { s1[tid] += s1[tid + off]; s2[tid] += s2[tid + off]; }
    __syncthreads();
  }
  if (tid == 0) {
    const float m   = s1[0] / (float)N;
    const float var = s2[0] / (float)N - m * m;
    stats[2 * c]     = m;
    stats[2 * c + 1] = rsqrtf(var + 1e-5f);
  }
}

// normalize: xn = (in - mean)*rstd  -> f32 and bf16 copies.  layout (N,C)
__global__ __launch_bounds__(256) void k_norm(const float* __restrict__ in,
                                              const float* __restrict__ stats,
                                              float* __restrict__ outf,
                                              __bf16* __restrict__ outb) {
  const long idx = (long)blockIdx.x * 256 + threadIdx.x;   // over N*C
  const int c = (int)(idx & 127);
  const float v = (in[idx] - stats[2 * c]) * stats[2 * c + 1];
  outf[idx] = v;
  outb[idx] = f2bf(v);
}

// final: skip[c][n] = inorm2(t2)[n][c] + x[c][n]   (skip is (C,N) in d_out)
__global__ __launch_bounds__(256) void k_final(const float* __restrict__ t2,
                                               const float* __restrict__ stats,
                                               const float* __restrict__ x,
                                               float* __restrict__ skip) {
  const long idx = (long)blockIdx.x * 256 + threadIdx.x;   // over C*N
  const int c = (int)(idx >> 15);
  const int n = (int)(idx & 32767);
  const float v = (t2[(size_t)n * 128 + c] - stats[2 * c]) * stats[2 * c + 1]
                + x[idx];
  skip[idx] = v;
}

// 2x2x2 maxpool: down (C,16,16,16) from skip (C,32,32,32)
__global__ __launch_bounds__(256) void k_pool(const float* __restrict__ skip,
                                              float* __restrict__ down) {
  const long idx = (long)blockIdx.x * 256 + threadIdx.x;   // over 128*4096
  const int oz = (int)(idx & 15);
  const int ow = (int)((idx >> 4) & 15);
  const int oh = (int)((idx >> 8) & 15);
  const int c  = (int)(idx >> 12);
  const float* s = skip + (((size_t)c * 32 + oh * 2) * 32 + ow * 2) * 32 + oz * 2;
  float m = -3.0e38f;
#pragma unroll
  for (int a = 0; a < 2; ++a)
#pragma unroll
    for (int b = 0; b < 2; ++b)
#pragma unroll
      for (int d = 0; d < 2; ++d)
        m = fmaxf(m, s[a * 1024 + b * 32 + d]);
  down[idx] = m;
}

// ---------------------------------------------------------------------------
// Host launcher
// ---------------------------------------------------------------------------
extern "C" void kernel_launch(void* const* d_in, const int* in_sizes, int n_in,
                              void* d_out, int out_size, void* d_ws, size_t ws_size,
                              hipStream_t stream) {
  (void)in_sizes; (void)n_in; (void)out_size; (void)ws_size;

  const float* x      = (const float*)d_in[0];
  const float* qkv_w  = (const float*)d_in[1];
  const float* qkv_b  = (const float*)d_in[2];
  const float* proj_w = (const float*)d_in[3];
  const float* proj_b = (const float*)d_in[4];
  const float* rpb    = (const float*)d_in[5];
  const float* w1     = (const float*)d_in[6];
  const float* b1     = (const float*)d_in[7];
  const float* w2     = (const float*)d_in[8];
  const float* b2     = (const float*)d_in[9];

  constexpr int C = 128, F = 512, O = 384, N = 32768;
  constexpr long DOWN = 128L * 16 * 16 * 16;   // 524288

  char* ws = (char*)d_ws;
  size_t off = 0;
  auto alloc = [&](size_t bytes) {
    char* p = ws + off;
    off += (bytes + 255) & ~(size_t)255;
    return p;
  };
  __bf16* WQ   = (__bf16*)alloc((size_t)O * C * 2);
  __bf16* WP   = (__bf16*)alloc((size_t)C * C * 2);
  __bf16* W1T  = (__bf16*)alloc((size_t)F * C * 2);
  __bf16* W2T  = (__bf16*)alloc((size_t)C * F * 2);
  float*  ST   = (float*) alloc((size_t)C * 2 * 4);
  __bf16* RB   = (__bf16*)alloc((size_t)N * C * 2);    // XB -> SAMB -> XNB
  char*   RA   =          alloc((size_t)N * F * 2);    // QKV (24MB) then HID (32MB)
  float*  SPROJ= (float*) alloc((size_t)N * C * 4);
  float*  XN   = (float*) alloc((size_t)N * C * 4);
  float*  MLP  = (float*) alloc((size_t)N * C * 4);    // becomes t2 (resid fused)

  __bf16* XB   = RB;  __bf16* SAMB = RB;  __bf16* XNB = RB;   // time-shared
  __bf16* QKV  = (__bf16*)RA;
  __bf16* HID  = (__bf16*)RA;

  float* down = (float*)d_out;
  float* skip = (float*)d_out + DOWN;

  // 1. precision conversion / weight transposes
  k_cvt_xT<<<(C * N) / 256, 256, 0, stream>>>(x, XB);
  k_cvt  <<<(O * C) / 256, 256, 0, stream>>>(qkv_w, WQ);
  k_cvt  <<<(C * C) / 256, 256, 0, stream>>>(proj_w, WP);
  k_cvt_T<<<(C * F) / 256, 256, 0, stream>>>(w1, W1T, C, F);   // (C,F)->(F,C)
  k_cvt_T<<<(F * C) / 256, 256, 0, stream>>>(w2, W2T, F, C);   // (F,C)->(C,F)

  // 2. qkv GEMM (bf16 out, fused bias)
  k_gemm<1, 0, 0, 128><<<dim3(N / 256, O / 16), 256, 0, stream>>>(
      WQ, XB, qkv_b, nullptr, QKV, O, N);

  // 3. neighborhood attention
  k_attn<<<(N * 8) / 256, 256, 0, stream>>>(QKV, rpb, SAMB);

  // 4. output projection (f32 out)
  k_gemm<0, 0, 0, 128><<<dim3(N / 256, C / 16), 256, 0, stream>>>(
      WP, SAMB, proj_b, nullptr, SPROJ, C, N);

  // 5. instance norm 1
  k_stats<<<C, 256, 0, stream>>>(SPROJ, ST, C, N);
  k_norm <<<(N * C) / 256, 256, 0, stream>>>(SPROJ, ST, XN, XNB);

  // 6. MLP: hidden = relu(xn*w1+b1) [bf16]; t2 = xn + hidden*w2 + b2 [f32]
  k_gemm<1, 1, 0, 128><<<dim3(N / 256, F / 16), 256, 0, stream>>>(
      W1T, XNB, b1, nullptr, HID, F, N);
  k_gemm<0, 0, 1, 512><<<dim3(N / 256, C / 16), 256, 0, stream>>>(
      W2T, HID, b2, XN, MLP, C, N);

  // 7. instance norm 2 + residual with x -> skip; then maxpool -> down
  k_stats<<<C, 256, 0, stream>>>(MLP, ST, C, N);
  k_final<<<(C * N) / 256, 256, 0, stream>>>(MLP, ST, x, skip);
  k_pool <<<(int)(DOWN / 256), 256, 0, stream>>>(skip, down);
}
